// MultiTaskModel_48180943126747
// MI455X (gfx1250) — matrix operand order repaired and checked by reading, not compiled, not run on hardware
//
#include <hip/hip_runtime.h>
#include <math.h>

typedef __bf16 bf16;
typedef __attribute__((ext_vector_type(16))) __bf16 v16bf;
typedef __attribute__((ext_vector_type(8)))  __bf16 v8bf;
typedef __attribute__((ext_vector_type(8)))  float  v8f;

constexpr int Bb   = 2;
constexpr int Ls   = 2048;
constexpr int Dm   = 384;
constexpr int Hh   = 4;
constexpr int Dh   = 96;       // Dm / Hh
constexpr int FF   = 1536;
constexpr int NLay = 2;
constexpr int NCn  = 10;
constexpr int NRr  = 1024;
constexpr int CCh  = 128;
constexpr int Mrows = Bb * Ls; // 4096

#define LN_EPS   1e-5f
#define BN_EPS   1e-5f
#define QK_SCALE 0.10206207261596575f   /* 1/sqrt(96) */
#define INV2SIG2 1.2437733222305893e-05f /* 1/(2*(401/2)^2) */

// ---------------- wave32 reductions ----------------
__device__ inline float wave_max_f(float v) {
  #pragma unroll
  for (int o = 16; o > 0; o >>= 1) v = fmaxf(v, __shfl_xor(v, o, 32));
  return v;
}
__device__ inline float wave_sum_f(float v) {
  #pragma unroll
  for (int o = 16; o > 0; o >>= 1) v += __shfl_xor(v, o, 32);
  return v;
}

// ---------------- weight fp32 [K][N] -> bf16 [N][K] ----------------
__global__ void convert_wT_kernel(const float* __restrict__ W, bf16* __restrict__ WT,
                                  int K, int N) {
  int id = blockIdx.x * blockDim.x + threadIdx.x;
  if (id >= K * N) return;
  int n = id / K, k = id % K;
  WT[id] = (bf16)W[(size_t)k * N + n];
}

// ---------------- conv(3,5,9) + BN + ReLU + concat + sin PE ----------------
__global__ void conv_pe_kernel(const float* __restrict__ x,
                               const float* __restrict__ cw0,
                               const float* __restrict__ cw1,
                               const float* __restrict__ cw2,
                               const float* __restrict__ cb,
                               const float* __restrict__ bg,
                               const float* __restrict__ bbet,
                               const float* __restrict__ bm,
                               const float* __restrict__ bv,
                               float* __restrict__ h) {
  int id = blockIdx.x * blockDim.x + threadIdx.x;
  if (id >= Bb * Ls * Dm) return;
  int d = id % Dm;
  int l = (id / Dm) % Ls;
  int b = id / (Dm * Ls);
  int ci = d / CCh;
  int c  = d % CCh;
  int Kz = (ci == 0) ? 3 : (ci == 1) ? 5 : 9;
  const float* w = ((ci == 0) ? cw0 : (ci == 1) ? cw1 : cw2) + c * Kz;
  int pad = Kz >> 1;
  float y = 0.0f;
  for (int k = 0; k < Kz; ++k) {
    int xi = l + k - pad;
    if (xi >= 0 && xi < Ls) y += w[k] * x[b * Ls + xi];
  }
  int pc = ci * CCh + c;
  y += cb[pc];
  y = (y - bm[pc]) * rsqrtf(bv[pc] + BN_EPS) * bg[pc] + bbet[pc];
  y = fmaxf(y, 0.0f);
  // sinusoidal PE
  int p = d >> 1;
  float freq = expf(-(float)(2 * p) * (9.210340371976184f / (float)Dm));
  float ang  = (float)l * freq;
  float pe   = (d & 1) ? cosf(ang) : sinf(ang);
  h[id] = y + pe;
}

// ---------------- LayerNorm (wave per row) -> bf16 ----------------
__global__ void layernorm_bf16_kernel(const float* __restrict__ X,
                                      const float* __restrict__ g,
                                      const float* __restrict__ beta,
                                      bf16* __restrict__ out) {
  int wid  = (blockIdx.x * blockDim.x + threadIdx.x) >> 5;
  int lane = threadIdx.x & 31;
  if (wid >= Mrows) return;
  const float* xp = X + (size_t)wid * Dm;
  float vals[Dm / 32];
  float s = 0.0f;
  #pragma unroll
  for (int r = 0; r < Dm / 32; ++r) { vals[r] = xp[lane + 32 * r]; s += vals[r]; }
  s = wave_sum_f(s);
  float mu = s / (float)Dm;
  float q = 0.0f;
  #pragma unroll
  for (int r = 0; r < Dm / 32; ++r) { float t = vals[r] - mu; q += t * t; }
  q = wave_sum_f(q);
  float rstd = rsqrtf(q / (float)Dm + LN_EPS);
  bf16* op = out + (size_t)wid * Dm;
  #pragma unroll
  for (int r = 0; r < Dm / 32; ++r) {
    int d = lane + 32 * r;
    op[d] = (bf16)((vals[r] - mu) * rstd * g[d] + beta[d]);
  }
}

// ---------------- WMMA bf16 GEMM: C[M][N] = A[M][K] * BT[N][K]^T ----------------
// 2x2 register-blocked: one 32x32 macro-tile (4 WMMA tiles) per wave.
// Requires K % 32 == 0, M % 32 == 0, N % 32 == 0.
__device__ inline void gemm_store_tile(v8f acc, int rbase, int cn, int N,
                                       const float* __restrict__ bias,
                                       const float* __restrict__ resid,
                                       float* __restrict__ outF,
                                       bf16* __restrict__ outB, int relu) {
  #pragma unroll
  for (int r = 0; r < 8; ++r) {
    size_t idx = (size_t)(rbase + r) * N + cn;
    float c = acc[r];
    if (bias)  c += bias[cn];
    if (resid) c += resid[idx];
    if (relu)  c = fmaxf(c, 0.0f);
    if (outF)  outF[idx] = c;
    if (outB)  outB[idx] = (bf16)c;
  }
}

__device__ inline v16bf frag_assemble(v8bf lo, v8bf hi) {
  v16bf f;
  #pragma unroll
  for (int e = 0; e < 8; ++e) { f[e] = lo[e]; f[e + 8] = hi[e]; }
  return f;
}

__global__ void gemm_wmma_bf16_kernel(const bf16* __restrict__ A,
                                      const bf16* __restrict__ BT,
                                      const float* __restrict__ bias,
                                      const float* __restrict__ resid,
                                      float* __restrict__ outF,
                                      bf16* __restrict__ outB,
                                      int M, int N, int K, int relu) {
  int wid = (blockIdx.x * blockDim.x + threadIdx.x) >> 5;
  int tilesN = N >> 5;                // 32-wide macro-tiles
  int total  = (M >> 5) * tilesN;
  if (wid >= total) return;
  int tm = wid / tilesN, tn = wid % tilesN;
  int lane = threadIdx.x & 31;
  int hsel = lane >> 4;               // lane half
  int l15  = lane & 15;
  // A fragment lane layout (16-bit A 16x32): lanes 0-15 K {0..7,16..23}, lanes 16-31 K {8..15,24..31}
  const bf16* ap0 = A  + (size_t)(tm * 32 + l15) * K + hsel * 8;
  const bf16* ap1 = ap0 + (size_t)16 * K;
  // B fragment lane layout (16-bit B 32x16): lanes 0-15 K {0..7,16..23}, lanes 16-31 K {8..15,24..31}
  const bf16* bp0 = BT + (size_t)(tn * 32 + l15) * K + hsel * 8;
  const bf16* bp1 = bp0 + (size_t)16 * K;
  v8f acc00 = {}, acc01 = {}, acc10 = {}, acc11 = {};
  for (int k0 = 0; k0 < K; k0 += 32) {
    v8bf a0lo = *(const v8bf*)(ap0 + k0);
    v8bf a0hi = *(const v8bf*)(ap0 + k0 + 16);
    v8bf a1lo = *(const v8bf*)(ap1 + k0);
    v8bf a1hi = *(const v8bf*)(ap1 + k0 + 16);
    v8bf b0lo = *(const v8bf*)(bp0 + k0);
    v8bf b0hi = *(const v8bf*)(bp0 + k0 + 16);
    v8bf b1lo = *(const v8bf*)(bp1 + k0);
    v8bf b1hi = *(const v8bf*)(bp1 + k0 + 16);
    __builtin_prefetch((const void*)(ap0 + k0 + 64), 0, 3);
    __builtin_prefetch((const void*)(ap1 + k0 + 64), 0, 3);
    __builtin_prefetch((const void*)(bp0 + k0 + 64), 0, 3);
    __builtin_prefetch((const void*)(bp1 + k0 + 64), 0, 3);
    v16bf a0 = frag_assemble(a0lo, a0hi);
    v16bf a1 = frag_assemble(a1lo, a1hi);
    v16bf b0 = frag_assemble(b0lo, b0hi);
    v16bf b1 = frag_assemble(b1lo, b1hi);
    acc00 = __builtin_amdgcn_wmma_f32_16x16x32_bf16(false, a0, false, b0, (short)0, acc00, false, false);
    acc01 = __builtin_amdgcn_wmma_f32_16x16x32_bf16(false, a0, false, b1, (short)0, acc01, false, false);
    acc10 = __builtin_amdgcn_wmma_f32_16x16x32_bf16(false, a1, false, b0, (short)0, acc10, false, false);
    acc11 = __builtin_amdgcn_wmma_f32_16x16x32_bf16(false, a1, false, b1, (short)0, acc11, false, false);
  }
  // C/D layout: element r -> row = base + (lane<16 ? 0 : 8) + r, col = tile_col + (lane&15)
  int cn0    = tn * 32 + l15;
  int cn1    = cn0 + 16;
  int rbase0 = tm * 32 + (hsel << 3);
  int rbase1 = rbase0 + 16;
  gemm_store_tile(acc00, rbase0, cn0, N, bias, resid, outF, outB, relu);
  gemm_store_tile(acc01, rbase0, cn1, N, bias, resid, outF, outB, relu);
  gemm_store_tile(acc10, rbase1, cn0, N, bias, resid, outF, outB, relu);
  gemm_store_tile(acc11, rbase1, cn1, N, bias, resid, outF, outB, relu);
}

// ---------------- mean of V over sequence (fallback when no score>0) ----------------
__global__ void meanv_kernel(const float* __restrict__ V, float* __restrict__ meanV) {
  int id = blockIdx.x * blockDim.x + threadIdx.x;
  if (id >= Bb * Hh * Dh) return;
  int e  = id % Dh;
  int hh = (id / Dh) % Hh;
  int b  = id / (Dh * Hh);
  float s = 0.0f;
  for (int j = 0; j < Ls; ++j) s += V[(size_t)(b * Ls + j) * Dm + hh * Dh + e];
  meanV[id] = s / (float)Ls;
}

// ---------------- sparse attention rows i>=1 (wave per (b,h,i)) ----------------
__global__ void attn_sparse_kernel(const float* __restrict__ Q,
                                   const float* __restrict__ Km,
                                   const float* __restrict__ V,
                                   const float* __restrict__ meanV,
                                   bf16* __restrict__ out) {
  int wid  = (blockIdx.x * blockDim.x + threadIdx.x) >> 5;
  int lane = threadIdx.x & 31;
  int rows = Bb * Hh * (Ls - 1);
  if (wid >= rows) return;
  int i  = wid % (Ls - 1) + 1;
  int bh = wid / (Ls - 1);
  int hh = bh % Hh;
  int b  = bh / Hh;
  int jlo = i - 5; if (jlo < 0) jlo = 0;
  int jhi = i + 5; if (jhi > Ls - 1) jhi = Ls - 1;
  int wc = jhi - jlo + 1;
  bool extra0 = (jlo > 0);          // global token j=0
  int nc = wc + (extra0 ? 1 : 0);   // <= 12
  bool active = lane < nc;
  int j = 0;
  if (active) j = extra0 ? ((lane == 0) ? 0 : (jlo + lane - 1)) : (jlo + lane);

  const float* qp = Q + (size_t)(b * Ls + i) * Dm + hh * Dh;
  float s = 0.0f;
  if (active) {
    const float* kp = Km + (size_t)(b * Ls + j) * Dm + hh * Dh;
    #pragma unroll 4
    for (int e = 0; e < Dh; ++e) s += qp[e] * kp[e];
    s *= QK_SCALE;
  }
  bool kept = active && (s > 0.0f);  // TAU = 0
  float a = -INFINITY;
  if (kept) {
    float dd = (float)abs(i - j);
    float bias = expf(-dd * dd * INV2SIG2) + (((abs(i - j)) % 401 == 0) ? 0.3f : 0.0f);
    a = s + bias;
  }
  float m = wave_max_f(a);
  bool anyKept = (m != -INFINITY);
  float p = kept ? expf(a - m) : 0.0f;
  float S = wave_sum_f(p);

  float o0 = 0.0f, o1 = 0.0f, o2 = 0.0f;
  for (int c = 0; c < nc; ++c) {
    float pc = __shfl(p, c, 32);
    int   jc = __shfl(j, c, 32);
    const float* vp = V + (size_t)(b * Ls + jc) * Dm + hh * Dh;
    o0 += pc * vp[lane];
    o1 += pc * vp[lane + 32];
    o2 += pc * vp[lane + 64];
  }
  if (anyKept) { float inv = 1.0f / S; o0 *= inv; o1 *= inv; o2 *= inv; }
  else {
    const float* mv = meanV + (size_t)(b * Hh + hh) * Dh;
    o0 = mv[lane]; o1 = mv[lane + 32]; o2 = mv[lane + 64];
  }
  bf16* op = out + (size_t)(b * Ls + i) * Dm + hh * Dh;
  op[lane]      = (bf16)o0;
  op[lane + 32] = (bf16)o1;
  op[lane + 64] = (bf16)o2;
}

// ---------------- dense attention row i==0 (block per (b,h)) ----------------
__global__ void attn_row0_kernel(const float* __restrict__ Q,
                                 const float* __restrict__ Km,
                                 const float* __restrict__ V,
                                 const float* __restrict__ meanV,
                                 bf16* __restrict__ out) {
  __shared__ float qs[Dh];
  __shared__ float sa[Ls];
  __shared__ float red[256];
  int t  = threadIdx.x;
  int hh = blockIdx.x % Hh;
  int b  = blockIdx.x / Hh;
  const float* qp = Q + (size_t)(b * Ls) * Dm + hh * Dh;
  if (t < Dh) qs[t] = qp[t];
  __syncthreads();
  float lmax = -INFINITY;
  for (int j = t; j < Ls; j += 256) {
    const float* kp = Km + (size_t)(b * Ls + j) * Dm + hh * Dh;
    float s = 0.0f;
    #pragma unroll 4
    for (int e = 0; e < Dh; ++e) s += qs[e] * kp[e];
    s *= QK_SCALE;
    float a = -INFINITY;
    if (s > 0.0f) {
      float dd = (float)j;
      float bias = expf(-dd * dd * INV2SIG2) + ((j % 401 == 0) ? 0.3f : 0.0f);
      a = s + bias;
    }
    sa[j] = a;
    lmax = fmaxf(lmax, a);
  }
  red[t] = lmax;
  __syncthreads();
  for (int sft = 128; sft > 0; sft >>= 1) {
    if (t < sft) red[t] = fmaxf(red[t], red[t + sft]);
    __syncthreads();
  }
  float m = red[0];
  __syncthreads();
  float lsum = 0.0f;
  for (int j = t; j < Ls; j += 256) {
    float pv = (sa[j] == -INFINITY) ? 0.0f : expf(sa[j] - m);
    sa[j] = pv;
    lsum += pv;
  }
  red[t] = lsum;
  __syncthreads();
  for (int sft = 128; sft > 0; sft >>= 1) {
    if (t < sft) red[t] += red[t + sft];
    __syncthreads();
  }
  float S = red[0];
  bool anyKept = (m != -INFINITY);
  if (t < Dh) {
    float o;
    if (anyKept) {
      o = 0.0f;
      for (int j = 0; j < Ls; ++j)
        o += sa[j] * V[(size_t)(b * Ls + j) * Dm + hh * Dh + t];
      o /= S;
    } else {
      o = meanV[(size_t)(b * Hh + hh) * Dh + t];
    }
    out[(size_t)(b * Ls) * Dm + hh * Dh + t] = (bf16)o;
  }
}

// ---------------- cls / reg heads on pooled token 0 ----------------
__global__ void heads_kernel(const float* __restrict__ h,
                             const float* __restrict__ cls_w, const float* __restrict__ cls_b,
                             const float* __restrict__ reg_w, const float* __restrict__ reg_b,
                             float* __restrict__ outp) {
  int id = blockIdx.x * blockDim.x + threadIdx.x;
  int total = Bb * (NCn + NRr);
  if (id >= total) return;
  int b = id / (NCn + NRr);
  int r = id % (NCn + NRr);
  const float* pooled = h + (size_t)(b * Ls) * Dm;
  if (r < NCn) {
    float s = cls_b[r];
    for (int e = 0; e < Dm; ++e) s += pooled[e] * cls_w[(size_t)e * NCn + r];
    outp[b * NCn + r] = s;
  } else {
    int rr = r - NCn;
    float s = reg_b[rr];
    for (int e = 0; e < Dm; ++e) s += pooled[e] * reg_w[(size_t)e * NRr + rr];
    outp[Bb * NCn + b * NRr + rr] = s;
  }
}

// ---------------- host orchestration ----------------
static inline void* ws_alloc(char*& p, size_t bytes) {
  void* r = (void*)p;
  p += (bytes + 255) & ~(size_t)255;
  return r;
}

static inline void launch_gemm(const bf16* A, const bf16* BT, const float* bias,
                               const float* resid, float* outF, bf16* outB,
                               int M, int N, int K, int relu, hipStream_t stream) {
  int tiles = (M / 32) * (N / 32);  // 32x32 macro-tile per wave
  int blocks = (tiles + 7) / 8;     // 8 waves (256 threads) per block
  gemm_wmma_bf16_kernel<<<blocks, 256, 0, stream>>>(A, BT, bias, resid, outF, outB,
                                                    M, N, K, relu);
}

extern "C" void kernel_launch(void* const* d_in, const int* in_sizes, int n_in,
                              void* d_out, int out_size, void* d_ws, size_t ws_size,
                              hipStream_t stream) {
  const float* x      = (const float*)d_in[0];
  const float* cw0    = (const float*)d_in[1];
  const float* cw1    = (const float*)d_in[2];
  const float* cw2    = (const float*)d_in[3];
  const float* cb     = (const float*)d_in[4];
  const float* bng    = (const float*)d_in[5];
  const float* bnb    = (const float*)d_in[6];
  const float* bnm    = (const float*)d_in[7];
  const float* bnv    = (const float*)d_in[8];
  const float* qw     = (const float*)d_in[9];
  const float* qbias  = (const float*)d_in[10];
  const float* kw     = (const float*)d_in[11];
  const float* kbias  = (const float*)d_in[12];
  const float* vw     = (const float*)d_in[13];
  const float* vbias  = (const float*)d_in[14];
  const float* ow     = (const float*)d_in[15];
  const float* obias  = (const float*)d_in[16];
  const float* ln1g   = (const float*)d_in[17];
  const float* ln1b   = (const float*)d_in[18];
  const float* w1     = (const float*)d_in[19];
  const float* b1     = (const float*)d_in[20];
  const float* w2     = (const float*)d_in[21];
  const float* b2     = (const float*)d_in[22];
  const float* ln2g   = (const float*)d_in[23];
  const float* ln2b   = (const float*)d_in[24];
  const float* cls_w  = (const float*)d_in[25];
  const float* cls_b  = (const float*)d_in[26];
  const float* reg_w  = (const float*)d_in[27];
  const float* reg_b  = (const float*)d_in[28];
  float* outp = (float*)d_out;

  char* wp = (char*)d_ws;
  float* h      = (float*)ws_alloc(wp, (size_t)Mrows * Dm * 4);
  bf16*  hn     = (bf16*) ws_alloc(wp, (size_t)Mrows * Dm * 2);
  float* Qb     = (float*)ws_alloc(wp, (size_t)Mrows * Dm * 4);
  float* Kb     = (float*)ws_alloc(wp, (size_t)Mrows * Dm * 4);
  float* Vb     = (float*)ws_alloc(wp, (size_t)Mrows * Dm * 4);
  bf16*  attnB  = (bf16*) ws_alloc(wp, (size_t)Mrows * Dm * 2);
  bf16*  tB     = (bf16*) ws_alloc(wp, (size_t)Mrows * FF * 2);
  float* meanV  = (float*)ws_alloc(wp, (size_t)Bb * Hh * Dh * 4);
  bf16*  qwT[NLay]; bf16* kwT[NLay]; bf16* vwT[NLay]; bf16* owT[NLay];
  bf16*  w1T[NLay]; bf16* w2T[NLay];
  for (int l = 0; l < NLay; ++l) {
    qwT[l] = (bf16*)ws_alloc(wp, (size_t)Dm * Dm * 2);
    kwT[l] = (bf16*)ws_alloc(wp, (size_t)Dm * Dm * 2);
    vwT[l] = (bf16*)ws_alloc(wp, (size_t)Dm * Dm * 2);
    owT[l] = (bf16*)ws_alloc(wp, (size_t)Dm * Dm * 2);
    w1T[l] = (bf16*)ws_alloc(wp, (size_t)Dm * FF * 2);
    w2T[l] = (bf16*)ws_alloc(wp, (size_t)FF * Dm * 2);
  }

  // 1) weight conversion (fp32 [K][N] -> bf16 [N][K])
  for (int l = 0; l < NLay; ++l) {
    int nDD = Dm * Dm, nDF = Dm * FF;
    convert_wT_kernel<<<(nDD + 255) / 256, 256, 0, stream>>>(qw + (size_t)l * nDD, qwT[l], Dm, Dm);
    convert_wT_kernel<<<(nDD + 255) / 256, 256, 0, stream>>>(kw + (size_t)l * nDD, kwT[l], Dm, Dm);
    convert_wT_kernel<<<(nDD + 255) / 256, 256, 0, stream>>>(vw + (size_t)l * nDD, vwT[l], Dm, Dm);
    convert_wT_kernel<<<(nDD + 255) / 256, 256, 0, stream>>>(ow + (size_t)l * nDD, owT[l], Dm, Dm);
    convert_wT_kernel<<<(nDF + 255) / 256, 256, 0, stream>>>(w1 + (size_t)l * nDF, w1T[l], Dm, FF);
    convert_wT_kernel<<<(nDF + 255) / 256, 256, 0, stream>>>(w2 + (size_t)l * nDF, w2T[l], FF, Dm);
  }

  // 2) conv stem + BN + ReLU + PE
  {
    int total = Bb * Ls * Dm;
    conv_pe_kernel<<<(total + 255) / 256, 256, 0, stream>>>(
        x, cw0, cw1, cw2, cb, bng, bnb, bnm, bnv, h);
  }

  // 3) transformer layers
  for (int l = 0; l < NLay; ++l) {
    layernorm_bf16_kernel<<<Mrows / 8, 256, 0, stream>>>(h, ln1g + l * Dm, ln1b + l * Dm, hn);

    launch_gemm(hn, qwT[l], qbias + l * Dm, nullptr, Qb, nullptr, Mrows, Dm, Dm, 0, stream);
    launch_gemm(hn, kwT[l], kbias + l * Dm, nullptr, Kb, nullptr, Mrows, Dm, Dm, 0, stream);
    launch_gemm(hn, vwT[l], vbias + l * Dm, nullptr, Vb, nullptr, Mrows, Dm, Dm, 0, stream);

    meanv_kernel<<<(Bb * Hh * Dh + 255) / 256, 256, 0, stream>>>(Vb, meanV);
    attn_row0_kernel<<<Bb * Hh, 256, 0, stream>>>(Qb, Kb, Vb, meanV, attnB);
    {
      int rows = Bb * Hh * (Ls - 1);
      attn_sparse_kernel<<<(rows + 7) / 8, 256, 0, stream>>>(Qb, Kb, Vb, meanV, attnB);
    }

    // O projection + residual into h
    launch_gemm(attnB, owT[l], obias + l * Dm, h, h, nullptr, Mrows, Dm, Dm, 0, stream);

    // FFN
    layernorm_bf16_kernel<<<Mrows / 8, 256, 0, stream>>>(h, ln2g + l * Dm, ln2b + l * Dm, hn);
    launch_gemm(hn, w1T[l], b1 + l * FF, nullptr, nullptr, tB, Mrows, FF, Dm, 1, stream);
    launch_gemm(tB, w2T[l], b2 + l * Dm, h, h, nullptr, Mrows, Dm, FF, 0, stream);
  }

  // 4) heads from pooled token 0
  {
    int total = Bb * (NCn + NRr);
    heads_kernel<<<(total + 255) / 256, 256, 0, stream>>>(h, cls_w, cls_b, reg_w, reg_b, outp);
  }
}
